// DeepGCN_82454782148693
// MI455X (gfx1250) — matrix operand order repaired and checked by reading, not compiled, not run on hardware
//
#include <hip/hip_runtime.h>
#include <hip/hip_bf16.h>

// ---------------------------------------------------------------------------
// DeepGCN forward for MI455X (gfx1250, wave32, WMMA).
// Dense GEMMs: v_wmma_f32_16x16x32_bf16 (bf16 in, f32 accum).
// Sparse aggregation: wave-per-edge float4 streaming + f32 global atomics
// (agg buffer is 51MB -> resident in 192MB L2).
// ---------------------------------------------------------------------------

#define GH 128          // hidden width
#define GHC 64          // classifier hidden
#define GL 4            // layers
#define C1 0.4f         // 1 - ALPHA - THETA
#define GALPHA 0.1f
#define GTHETA 0.5f
#define LDA 136         // LDS A stride in bf16 elems (272B, 16B aligned rows)

#if __has_builtin(__builtin_amdgcn_sched_barrier)
#define SCHED_FENCE() __builtin_amdgcn_sched_barrier(0)
#else
#define SCHED_FENCE()
#endif

typedef __attribute__((ext_vector_type(4)))  __bf16 v4bf;
typedef __attribute__((ext_vector_type(8)))  __bf16 v8bf;
typedef __attribute__((ext_vector_type(16))) __bf16 v16bf;
typedef __attribute__((ext_vector_type(8)))  float  v8f;

union Frag16 { v16bf v; v8bf h[2]; };

static __device__ inline __bf16 f2bf(float f) {
    // round-to-nearest-even f32 -> bf16 via bit twiddle (no codegen surprises)
    unsigned u = __builtin_bit_cast(unsigned, f);
    unsigned r = (u + 0x7FFFu + ((u >> 16) & 1u)) >> 16;
    return __builtin_bit_cast(__bf16, (unsigned short)r);
}

// ------------------------- degree / normalization --------------------------

__global__ void k_deg_init(float* deg, int n) {
    int i = blockIdx.x * 256 + threadIdx.x;
    if (i < n) deg[i] = 1.0f;                       // self-loop
}

__global__ void k_deg_edges(float* deg, const int* __restrict__ dst, int e) {
    int i = blockIdx.x * 256 + threadIdx.x;
    if (i < e) atomicAdd(&deg[dst[i]], 1.0f);
}

__global__ void k_dinv(float* deg, int n) {
    int i = blockIdx.x * 256 + threadIdx.x;
    if (i < n) deg[i] = rsqrtf(fmaxf(deg[i], 1.0f));  // in place: deg -> dinv
}

// ------------------------- weight convert+transpose ------------------------
// W is [K, Nc] row-major f32; Bt is [Nc, K] bf16 (B-fragment friendly).

__global__ void k_convert_wt(const float* __restrict__ W, __bf16* __restrict__ Bt,
                             int K, int Nc) {
    int idx = blockIdx.x * 256 + threadIdx.x;
    if (idx < K * Nc) {
        int k = idx / Nc, nn = idx % Nc;
        Bt[(size_t)nn * K + k] = f2bf(W[idx]);
    }
}

// ------------------------------- WMMA GEMM ---------------------------------
// C[nRows, nCols] = A[nRows,128] @ W[128,nCols] + bias (optional relu)
// A: f32 global, staged to LDS as bf16. Bt: bf16 [nCols,128] (transposed,
// L2-resident). bias: always a valid pointer (zero scratch when unused).
// Block = 256 threads = 8 wave32; wave w owns rows [16w,16w+16) of a
// 128-row block, loops over 16-col tiles, K = 4 x wmma 16x16x32.

template <int RELU>
__global__ __launch_bounds__(256) void k_gemm128(
    const float* __restrict__ A, const __bf16* __restrict__ Bt,
    const float* __restrict__ bias, float* __restrict__ C,
    int nRows, int nCols)
{
    __shared__ __bf16 sA[128 * LDA];                 // 34,816 B
    const int tid = threadIdx.x;
    const int blockRow0 = blockIdx.x * 128;
    const bool fullTile = (blockRow0 + 128 <= nRows);

    // Stage A tile: f32 -> bf16, 128x128, rows beyond nRows zero-filled.
    for (int chunk = tid; chunk < (128 * 128) / 4; chunk += 256) {
        int row  = chunk >> 5;
        int coff = (chunk & 31) << 2;
        int gRow = blockRow0 + row;
        float4 v = make_float4(0.f, 0.f, 0.f, 0.f);
        if (gRow < nRows) v = *(const float4*)(A + (size_t)gRow * GH + coff);
        v4bf t; t[0] = f2bf(v.x); t[1] = f2bf(v.y); t[2] = f2bf(v.z); t[3] = f2bf(v.w);
        *(v4bf*)&sA[row * LDA + coff] = t;
    }
    __syncthreads();

    const int wave      = tid >> 5;                  // wave32
    const int lane      = tid & 31;
    const int laneLo    = lane & 15;
    const int laneHiSel = (lane >> 4) & 1;
    const int rowOff    = (wave * 16 + laneLo) * LDA;
    const int nct       = nCols >> 4;

    // A fragments (ISA 7.12.2: lanes 0-15 rows M, K 0..7 & 16..23; lanes
    // 16-31 same rows, K 8..15 & 24..31) — reused across all col tiles.
    Frag16 fa[4];
#pragma unroll
    for (int kb = 0; kb < 4; ++kb) {
        int base0 = kb * 32 + laneHiSel * 8;
        fa[kb].h[0] = *(const v8bf*)&sA[rowOff + base0];
        fa[kb].h[1] = *(const v8bf*)&sA[rowOff + base0 + 16];
    }

    for (int ct = 0; ct < nct; ++ct) {
        int col = ct * 16 + laneLo;

        // Issue ALL B-fragment loads first; the sched fence stops the
        // scheduler from re-sinking them, so the waits become partial
        // (loadcnt<=6/4/2/0) and WMMA k overlaps the loads for k+1..3.
        Frag16 fb[4];
#pragma unroll
        for (int kb = 0; kb < 4; ++kb) {
            const __bf16* bp = Bt + (size_t)col * GH + kb * 32 + laneHiSel * 8;
            fb[kb].h[0] = *(const v8bf*)bp;
            fb[kb].h[1] = *(const v8bf*)(bp + 16);
        }
        float bv = bias[col];
        SCHED_FENCE();
        v8f acc = {};
#pragma unroll
        for (int kb = 0; kb < 4; ++kb) {
            acc = __builtin_amdgcn_wmma_f32_16x16x32_bf16(
                false, fa[kb].v, false, fb[kb].v, (short)0, acc, false, false);
        }

        int rbase = blockRow0 + wave * 16 + laneHiSel * 8;   // C: M = r + 8*lane[4]
        if (fullTile) {
            // Uniform fast path: no per-row exec-mask surgery.
#pragma unroll
            for (int r = 0; r < 8; ++r) {
                float v = acc[r] + bv;
                if (RELU) v = fmaxf(v, 0.0f);
                C[(size_t)(rbase + r) * nCols + col] = v;
            }
        } else {
#pragma unroll
            for (int r = 0; r < 8; ++r) {
                int row = rbase + r;
                if (row < nRows) {
                    float v = acc[r] + bv;
                    if (RELU) v = fmaxf(v, 0.0f);
                    C[(size_t)row * nCols + col] = v;
                }
            }
        }
    }
}

// --------------------------- sparse aggregation ----------------------------

// agg[i][c] = h[i][c] * dinv[i]^2 + conv_b[c]   (self-loop + bias seed)
__global__ void k_agg_init(const float* __restrict__ h, const float* __restrict__ dinv,
                           const float* __restrict__ cb, float* __restrict__ agg,
                           long long total) {
    long long i = (long long)blockIdx.x * 256 + threadIdx.x;
    if (i >= total) return;
    int node = (int)(i >> 7);
    int c = (int)(i & 127);
    float di = dinv[node];
    agg[i] = h[i] * di * di + cb[c];
}

// one wave32 per edge: 4 floats/lane gather, scale, scatter-add
__global__ __launch_bounds__(256) void k_edge_agg(
    const float* __restrict__ h, const float* __restrict__ dinv,
    const int* __restrict__ src, const int* __restrict__ dst,
    float* __restrict__ agg, int e)
{
    int eid = blockIdx.x * 8 + (threadIdx.x >> 5);
    if (eid >= e) return;
    int lane = threadIdx.x & 31;
    int s = src[eid], d = dst[eid];
    float nrm = dinv[s] * dinv[d];
    __builtin_prefetch(agg + (size_t)d * GH, 0, 3);   // global_prefetch_b8
    float4 v = *(const float4*)(h + (size_t)s * GH + lane * 4);
    float* ad = agg + (size_t)d * GH + lane * 4;
    atomicAdd(ad + 0, v.x * nrm);
    atomicAdd(ad + 1, v.y * nrm);
    atomicAdd(ad + 2, v.z * nrm);
    atomicAdd(ad + 3, v.w * nrm);
}

// ------------------------------ batchnorm ----------------------------------

__global__ void k_zero(float* p, int n) {
    int i = blockIdx.x * 256 + threadIdx.x;
    if (i < n) p[i] = 0.0f;
}

// stats[c] = sum over rows, stats[128+c] = sumsq
__global__ void k_bn_stats(const float* __restrict__ y, float* __restrict__ stats,
                           int nRows) {
    int col = threadIdx.x & 127;
    int sub = threadIdx.x >> 7;                     // 0/1: two row streams
    float s = 0.f, sq = 0.f;
    for (int r = blockIdx.x * 2 + sub; r < nRows; r += gridDim.x * 2) {
        float v = y[(size_t)r * GH + col];
        s += v; sq += v * v;
    }
    __shared__ float ls[256], lq[256];
    ls[threadIdx.x] = s; lq[threadIdx.x] = sq;
    __syncthreads();
    if (sub == 0) {
        s  += ls[128 + col];
        sq += lq[128 + col];
        atomicAdd(&stats[col], s);
        atomicAdd(&stats[128 + col], sq);
    }
}

// bnp[c] = scale, bnp[128+c] = shift
__global__ void k_bn_final(const float* __restrict__ stats, const float* __restrict__ g,
                           const float* __restrict__ b, float* __restrict__ bnp,
                           float invN) {
    int c = threadIdx.x;
    float mean = stats[c] * invN;
    float var  = stats[128 + c] * invN - mean * mean;
    float inv  = rsqrtf(var + 1e-5f);
    float sc   = inv * g[c];
    bnp[c]        = sc;
    bnp[128 + c]  = b[c] - mean * sc;
}

// x_next = relu(C1*(y*scale+shift) + ALPHA*x0 + THETA*x_prev)
__global__ void k_epilogue(const float* __restrict__ y, const float* __restrict__ bnp,
                           const float* __restrict__ x0, const float* __restrict__ xp,
                           float* __restrict__ xn, long long total) {
    long long i = (long long)blockIdx.x * 256 + threadIdx.x;
    if (i >= total) return;
    int c = (int)(i & 127);
    float bnv = y[i] * bnp[c] + bnp[128 + c];
    float v = C1 * bnv + GALPHA * x0[i] + GTHETA * xp[i];
    xn[i] = fmaxf(v, 0.0f);
}

// ------------------------------ classifier tail ----------------------------
// out[i, 0..1] = y1[i,:64] @ W2[64,2] + b2 — tiny, plain VALU.
__global__ void k_cls2(const float* __restrict__ y1, const float* __restrict__ W2,
                       const float* __restrict__ b2, float* __restrict__ out,
                       int nRows) {
    int i = blockIdx.x * 256 + threadIdx.x;
    if (i >= nRows) return;
    const float4* row = (const float4*)(y1 + (size_t)i * GHC);
    float a0 = b2[0], a1 = b2[1];
#pragma unroll 4
    for (int kk = 0; kk < 16; ++kk) {
        float4 v = row[kk];
        int k = kk * 4;
        a0 += v.x * W2[(k+0)*2+0] + v.y * W2[(k+1)*2+0] + v.z * W2[(k+2)*2+0] + v.w * W2[(k+3)*2+0];
        a1 += v.x * W2[(k+0)*2+1] + v.y * W2[(k+1)*2+1] + v.z * W2[(k+2)*2+1] + v.w * W2[(k+3)*2+1];
    }
    out[(size_t)i * 2 + 0] = a0;
    out[(size_t)i * 2 + 1] = a1;
}

// ------------------------------- launcher ----------------------------------

extern "C" void kernel_launch(void* const* d_in, const int* in_sizes, int n_in,
                              void* d_out, int out_size, void* d_ws, size_t ws_size,
                              hipStream_t stream) {
    (void)n_in; (void)out_size; (void)ws_size;

    const float* x      = (const float*)d_in[0];
    const int*   ei     = (const int*)  d_in[1];
    const float* proj_W = (const float*)d_in[2];
    const float* proj_b = (const float*)d_in[3];
    const float* conv_W = (const float*)d_in[4];
    const float* conv_b = (const float*)d_in[5];
    const float* lin_W  = (const float*)d_in[6];
    const float* lin_b  = (const float*)d_in[7];
    const float* bn_g   = (const float*)d_in[8];
    const float* bn_b   = (const float*)d_in[9];
    const float* clsW1  = (const float*)d_in[10];
    const float* clsb1  = (const float*)d_in[11];
    const float* clsW2  = (const float*)d_in[12];
    const float* clsb2  = (const float*)d_in[13];
    float* out = (float*)d_out;

    const int N = in_sizes[0] / GH;          // 100000
    const int E = in_sizes[1] / 2;           // 1600000
    const int* src = ei;
    const int* dst = ei + E;

    // ---- workspace carve-up (256B aligned slices) ----
    char* w = (char*)d_ws;
    auto alloc = [&](size_t bytes) -> char* {
        char* p = w; w += (bytes + 255) & ~(size_t)255; return p;
    };
    const size_t NH = (size_t)N * GH;
    float*  dinv  = (float*)alloc((size_t)N * 4);
    float*  x0    = (float*)alloc(NH * 4);
    float*  xA    = (float*)alloc(NH * 4);
    float*  xB    = (float*)alloc(NH * 4);
    float*  hbuf  = (float*)alloc(NH * 4);       // conv output, then lin output
    float*  agg   = (float*)alloc(NH * 4);       // aggregation / classifier y1
    float*  stats = (float*)alloc(256 * 4);
    float*  bnp   = (float*)alloc(256 * 4);
    float*  zbias = (float*)alloc(128 * 4);      // zero bias for bias-less GEMMs
    __bf16* wproj = (__bf16*)alloc((size_t)GH * GH * 2);
    __bf16* wconv = (__bf16*)alloc((size_t)GL * GH * GH * 2);
    __bf16* wlin  = (__bf16*)alloc((size_t)GL * GH * GH * 2);
    __bf16* wcls1 = (__bf16*)alloc((size_t)GHC * GH * 2);

    const dim3 B256(256);
    const int gN    = (N + 255) / 256;
    const int gE    = (E + 255) / 256;
    const int gNH   = (int)((NH + 255) / 256);
    const int gGemm = (N + 127) / 128;
    const int gWt   = (GH * GH + 255) / 256;

    // ---- degree normalization (self-loops included) ----
    k_deg_init <<<gN, B256, 0, stream>>>(dinv, N);
    k_deg_edges<<<gE, B256, 0, stream>>>(dinv, dst, E);
    k_dinv     <<<gN, B256, 0, stream>>>(dinv, N);
    k_zero     <<<1, B256, 0, stream>>>(zbias, 128);

    // ---- weights -> transposed bf16 ----
    k_convert_wt<<<gWt, B256, 0, stream>>>(proj_W, wproj, GH, GH);
    for (int i = 0; i < GL; ++i) {
        k_convert_wt<<<gWt, B256, 0, stream>>>(conv_W + (size_t)i * GH * GH,
                                               wconv + (size_t)i * GH * GH, GH, GH);
        k_convert_wt<<<gWt, B256, 0, stream>>>(lin_W + (size_t)i * GH * GH,
                                               wlin + (size_t)i * GH * GH, GH, GH);
    }
    k_convert_wt<<<(GH * GHC + 255) / 256, B256, 0, stream>>>(clsW1, wcls1, GH, GHC);

    // ---- projection: x0 = relu(x @ proj_W + proj_b) ----
    k_gemm128<1><<<gGemm, B256, 0, stream>>>(x, wproj, proj_b, x0, N, GH);

    // ---- GCNII layers ----
    float* pingpong[2] = { xA, xB };
    const float* xcur = x0;
    for (int i = 0; i < GL; ++i) {
        float* xnext = pingpong[i & 1];
        // h = x @ conv_W[i]
        k_gemm128<0><<<gGemm, B256, 0, stream>>>(xcur, wconv + (size_t)i * GH * GH,
                                                 zbias, hbuf, N, GH);
        // agg = self-loops + bias, then edge scatter-adds
        k_agg_init<<<gNH, B256, 0, stream>>>(hbuf, dinv, conv_b + (size_t)i * GH,
                                             agg, (long long)NH);
        k_edge_agg<<<(E + 7) / 8, B256, 0, stream>>>(hbuf, dinv, src, dst, agg, E);
        // y = agg @ lin_W[i] + lin_b[i]   (into hbuf)
        k_gemm128<0><<<gGemm, B256, 0, stream>>>(agg, wlin + (size_t)i * GH * GH,
                                                 lin_b + (size_t)i * GH, hbuf, N, GH);
        // batchnorm stats + fused scale/shift
        k_zero    <<<1, B256, 0, stream>>>(stats, 256);
        k_bn_stats<<<512, B256, 0, stream>>>(hbuf, stats, N);
        k_bn_final<<<1, dim3(128), 0, stream>>>(stats, bn_g + (size_t)i * GH,
                                                bn_b + (size_t)i * GH, bnp,
                                                1.0f / (float)N);
        // residual + relu
        k_epilogue<<<gNH, B256, 0, stream>>>(hbuf, bnp, x0, xcur, xnext,
                                             (long long)NH);
        xcur = xnext;
    }

    // ---- classifier ----
    k_gemm128<1><<<gGemm, B256, 0, stream>>>(xcur, wcls1, clsb1, agg, N, GHC);
    k_cls2<<<gN, B256, 0, stream>>>(agg, clsW2, clsb2, out, N);
}
